// FastBinaryLinear_85521388798409
// MI455X (gfx1250) — matrix operand order repaired and checked by reading, not compiled, not run on hardware
//
#include <hip/hip_runtime.h>

// ---------------------------------------------------------------------------
// FastBinaryLinear: y = x @ (sign(W) * scale)^T
//   x: [8192, 4096] f32, W: [4096, 4096] f32, scale: [4096] f32
//
// Plan:
//   pass 1: x -> (x_hi, x_lo) bf16 split (near-fp32 accuracy; sign(W) is
//           exact in bf16), W -> sign bf16. Stored in workspace; the whole
//           converted working set (~167 MB) fits in the 192 MB L2.
//   pass 2: bf16 WMMA GEMM, double-buffered LDS staging via
//           GLOBAL_LOAD_ASYNC_TO_LDS_B128 (ASYNCcnt).
// Fallback (small ws): round-1 fused convert-in-kernel GEMM.
// ---------------------------------------------------------------------------

typedef __bf16 bf16;
typedef __attribute__((ext_vector_type(16))) __bf16 v16bf;
typedef __attribute__((ext_vector_type(8)))  __bf16 v8bf;
typedef __attribute__((ext_vector_type(4)))  __bf16 v4bf;
typedef __attribute__((ext_vector_type(8)))  float  v8f;
typedef __attribute__((ext_vector_type(4)))  int    v4i;

#define IN_F   4096
#define OUT_F  4096
#define TOKENS 8192

#define BM 64     // block tile M (tokens)
#define BN 128    // block tile N (out features)
#define BK 32     // K step (one bf16 WMMA deep)
#define LDA 40    // padded LDS row stride in halves (80 B; 16B-aligned chunks)
#define LDB 40
#define NK (IN_F / BK)

static_assert(TOKENS % BM == 0, "");
static_assert(OUT_F  % BN == 0, "");
static_assert(IN_F   % BK == 0, "");

#if defined(__has_builtin)
#  if __has_builtin(__builtin_amdgcn_global_load_async_to_lds_b128)
#    define HAVE_ASYNC_LDS 1
#  endif
#  if __has_builtin(__builtin_amdgcn_s_wait_asynccnt)
#    define HAVE_WAIT_ASYNC 1
#  endif
#endif

__device__ __forceinline__ bf16 f32_to_bf16_rne(float f) {
  unsigned u = __builtin_bit_cast(unsigned, f);
  unsigned r = u + 0x7FFFu + ((u >> 16) & 1u);
  unsigned short h = (unsigned short)(r >> 16);
  return __builtin_bit_cast(bf16, h);
}

__device__ __forceinline__ float bf16_to_f32(bf16 b) {
  unsigned short h = __builtin_bit_cast(unsigned short, b);
  unsigned u = ((unsigned)h) << 16;
  return __builtin_bit_cast(float, u);
}

// Build a 16-half fragment from two 16-byte LDS chunks.
__device__ __forceinline__ v16bf frag16(const bf16* p0, const bf16* p1) {
  v8bf a = *(const v8bf*)p0;
  v8bf b = *(const v8bf*)p1;
  v16bf r;
#pragma unroll
  for (int i = 0; i < 8; ++i) { r[i] = a[i]; r[i + 8] = b[i]; }
  return r;
}

// 16-byte global -> LDS copy; async (no VGPR round trip) when available.
// Builtin signature (from hipcc diagnostic): (v4i AS1*, v4i AS3*, imm, imm).
__device__ __forceinline__ void copy_b128_to_lds(const bf16* g, bf16* l) {
#if defined(HAVE_ASYNC_LDS)
  typedef __attribute__((address_space(1))) v4i* gp_t;
  typedef __attribute__((address_space(3))) v4i* lp_t;
  __builtin_amdgcn_global_load_async_to_lds_b128(
      (gp_t)(const void*)g, (lp_t)(void*)l, 0, 0);
#else
  *(v8bf*)l = *(const v8bf*)g;
#endif
}

__device__ __forceinline__ void wait_async_pending4() {
#if defined(HAVE_ASYNC_LDS)
#  if defined(HAVE_WAIT_ASYNC)
  __builtin_amdgcn_s_wait_asynccnt(4);
#  else
  asm volatile("s_wait_asynccnt 0x4" ::: "memory");
#  endif
#endif
}

__device__ __forceinline__ void wait_async_all() {
#if defined(HAVE_ASYNC_LDS)
#  if defined(HAVE_WAIT_ASYNC)
  __builtin_amdgcn_s_wait_asynccnt(0);
#  else
  asm volatile("s_wait_asynccnt 0x0" ::: "memory");
#  endif
#endif
}

// ---------------------------------------------------------------------------
// Pass 1 kernels: one-time conversion into workspace.
// ---------------------------------------------------------------------------
__global__ __launch_bounds__(256)
void convert_x_hilo(const float* __restrict__ x,
                    bf16* __restrict__ xhi, bf16* __restrict__ xlo) {
  size_t i4 = (size_t)blockIdx.x * blockDim.x + threadIdx.x;   // float4 index
  const float4 v = *(const float4*)(x + i4 * 4);
  float f[4] = {v.x, v.y, v.z, v.w};
  v4bf hi, lo;
#pragma unroll
  for (int j = 0; j < 4; ++j) {
    bf16 h = f32_to_bf16_rne(f[j]);
    hi[j] = h;
    lo[j] = f32_to_bf16_rne(f[j] - bf16_to_f32(h));
  }
  *(v4bf*)(xhi + i4 * 4) = hi;
  *(v4bf*)(xlo + i4 * 4) = lo;
}

__global__ __launch_bounds__(256)
void convert_w_sign(const float* __restrict__ w, bf16* __restrict__ wsg) {
  size_t i4 = (size_t)blockIdx.x * blockDim.x + threadIdx.x;
  const float4 v = *(const float4*)(w + i4 * 4);
  float f[4] = {v.x, v.y, v.z, v.w};
  v4bf s;
#pragma unroll
  for (int j = 0; j < 4; ++j) {
    unsigned short e = (f[j] > 0.0f) ? (unsigned short)0x3F80
                     : ((f[j] < 0.0f) ? (unsigned short)0xBF80 : (unsigned short)0);
    s[j] = __builtin_bit_cast(bf16, e);
  }
  *(v4bf*)(wsg + i4 * 4) = s;
}

// ---------------------------------------------------------------------------
// Pass 2: bf16 WMMA GEMM with double-buffered async LDS staging.
// Per stage (per wave): 4 async-load instructions (1 A-hi, 1 A-lo, 2 B).
// ---------------------------------------------------------------------------
__device__ __forceinline__ void stage_tiles(
    const bf16* __restrict__ xhi, const bf16* __restrict__ xlo,
    const bf16* __restrict__ wsg,
    bf16* sAh, bf16* sAl, bf16* sB,
    int mBase, int nBase, int k0, int tid) {
  // A tiles: BM x BK halves = 256 16B-chunks each; one chunk/thread.
  {
    int c = tid;
    int row = c >> 2, col8 = (c & 3) << 3;          // 4 chunks per 32-half row
    size_t goff = (size_t)(mBase + row) * IN_F + k0 + col8;
    copy_b128_to_lds(xhi + goff, sAh + row * LDA + col8);
    copy_b128_to_lds(xlo + goff, sAl + row * LDA + col8);
  }
  // B tile: BN x BK halves = 512 chunks; two chunks/thread.
#pragma unroll
  for (int it = 0; it < 2; ++it) {
    int c = it * 256 + tid;
    int row = c >> 2, col8 = (c & 3) << 3;
    size_t goff = (size_t)(nBase + row) * IN_F + k0 + col8;
    copy_b128_to_lds(wsg + goff, sB + row * LDB + col8);
  }
}

__global__ __launch_bounds__(256, 2)
void fbl_wmma_gemm_async(const bf16* __restrict__ xhi,
                         const bf16* __restrict__ xlo,
                         const bf16* __restrict__ wsg,
                         const float* __restrict__ scale,
                         float* __restrict__ out) {
  __shared__ __align__(16) bf16 sAh[2][BM * LDA];
  __shared__ __align__(16) bf16 sAl[2][BM * LDA];
  __shared__ __align__(16) bf16 sB [2][BN * LDB];

  const int tid   = threadIdx.x;
  const int lane  = tid & 31;
  const int wave  = tid >> 5;
  const int wm    = wave & 1;      // 2 waves along M
  const int wn    = wave >> 1;     // 4 waves along N
  const int mBase = blockIdx.y * BM;
  const int nBase = blockIdx.x * BN;
  const int l16   = lane & 15;
  const int lhalf = lane >> 4;

  v8f acc[2][2] = {};

  // Prologue: stage k=0 into buffer 0.
  stage_tiles(xhi, xlo, wsg, sAh[0], sAl[0], sB[0], mBase, nBase, 0, tid);

  for (int k = 0; k < NK; ++k) {
    const int cur = k & 1;
    if (k + 1 < NK) {
      stage_tiles(xhi, xlo, wsg, sAh[cur ^ 1], sAl[cur ^ 1], sB[cur ^ 1],
                  mBase, nBase, (k + 1) * BK, tid);
      wait_async_pending4();       // drain stage k (in-order), keep k+1 in flight
    } else {
      wait_async_all();
    }
    __syncthreads();               // all waves' stage-k data visible in LDS

    // Fragments (ISA 7.12.2 16-bit layouts).
    v16bf ah[2], al[2], bb[2];
#pragma unroll
    for (int mi = 0; mi < 2; ++mi) {
      const bf16* ph = sAh[cur] + (wm * 32 + mi * 16 + l16) * LDA + lhalf * 8;
      ah[mi] = frag16(ph, ph + 16);
      const bf16* pl = sAl[cur] + (wm * 32 + mi * 16 + l16) * LDA + lhalf * 8;
      al[mi] = frag16(pl, pl + 16);
    }
#pragma unroll
    for (int ni = 0; ni < 2; ++ni) {
      const bf16* pb = sB[cur] + (wn * 32 + ni * 16 + l16) * LDB + lhalf * 8;
      bb[ni] = frag16(pb, pb + 16);
    }

#pragma unroll
    for (int mi = 0; mi < 2; ++mi) {
#pragma unroll
      for (int ni = 0; ni < 2; ++ni) {
        acc[mi][ni] = __builtin_amdgcn_wmma_f32_16x16x32_bf16(
            false, ah[mi], false, bb[ni], (short)0, acc[mi][ni], false, false);
        acc[mi][ni] = __builtin_amdgcn_wmma_f32_16x16x32_bf16(
            false, al[mi], false, bb[ni], (short)0, acc[mi][ni], false, false);
      }
    }
    __syncthreads();               // reads done before buffer is restaged
  }

#pragma unroll
  for (int ni = 0; ni < 2; ++ni) {
    const int col = nBase + wn * 32 + ni * 16 + l16;
    const float s = scale[col];
#pragma unroll
    for (int mi = 0; mi < 2; ++mi) {
      const int rbase = mBase + wm * 32 + mi * 16 + (lhalf ? 8 : 0);
#pragma unroll
      for (int r = 0; r < 8; ++r) {
        out[(size_t)(rbase + r) * OUT_F + col] = acc[mi][ni][r] * s;
      }
    }
  }
}

// ---------------------------------------------------------------------------
// Fallback: fused kernel (convert during staging, no workspace).
// ---------------------------------------------------------------------------
__global__ __launch_bounds__(256, 2)
void fbl_wmma_gemm_fused(const float* __restrict__ x,
                         const float* __restrict__ w,
                         const float* __restrict__ scale,
                         float* __restrict__ out) {
  __shared__ __align__(16) bf16 sAh[BM * LDA];
  __shared__ __align__(16) bf16 sAl[BM * LDA];
  __shared__ __align__(16) bf16 sB [BN * LDB];

  const int tid   = threadIdx.x;
  const int lane  = tid & 31;
  const int wave  = tid >> 5;
  const int wm    = wave & 1;
  const int wn    = wave >> 1;
  const int mBase = blockIdx.y * BM;
  const int nBase = blockIdx.x * BN;
  const int l16   = lane & 15;
  const int lhalf = lane >> 4;

  v8f acc[2][2] = {};

  for (int k0 = 0; k0 < IN_F; k0 += BK) {
#pragma unroll
    for (int it = 0; it < (BM * BK) / (256 * 4); ++it) {
      int idx = it * 256 + tid;
      int row = idx >> 3;
      int c4  = (idx & 7) << 2;
      const float4 v = *(const float4*)(x + (size_t)(mBase + row) * IN_F + k0 + c4);
      float f[4] = {v.x, v.y, v.z, v.w};
      v4bf hi, lo;
#pragma unroll
      for (int j = 0; j < 4; ++j) {
        bf16 h = f32_to_bf16_rne(f[j]);
        hi[j] = h;
        lo[j] = f32_to_bf16_rne(f[j] - bf16_to_f32(h));
      }
      *(v4bf*)(sAh + row * LDA + c4) = hi;
      *(v4bf*)(sAl + row * LDA + c4) = lo;
    }
#pragma unroll
    for (int it = 0; it < (BN * BK) / (256 * 4); ++it) {
      int idx = it * 256 + tid;
      int row = idx >> 3;
      int c4  = (idx & 7) << 2;
      const float4 v = *(const float4*)(w + (size_t)(nBase + row) * IN_F + k0 + c4);
      float f[4] = {v.x, v.y, v.z, v.w};
      v4bf s;
#pragma unroll
      for (int j = 0; j < 4; ++j) {
        unsigned short e = (f[j] > 0.0f) ? (unsigned short)0x3F80
                         : ((f[j] < 0.0f) ? (unsigned short)0xBF80 : (unsigned short)0);
        s[j] = __builtin_bit_cast(bf16, e);
      }
      *(v4bf*)(sB + row * LDB + c4) = s;
    }
    __syncthreads();

    v16bf ah[2], al[2], bb[2];
#pragma unroll
    for (int mi = 0; mi < 2; ++mi) {
      const bf16* ph = sAh + (wm * 32 + mi * 16 + l16) * LDA + lhalf * 8;
      ah[mi] = frag16(ph, ph + 16);
      const bf16* pl = sAl + (wm * 32 + mi * 16 + l16) * LDA + lhalf * 8;
      al[mi] = frag16(pl, pl + 16);
    }
#pragma unroll
    for (int ni = 0; ni < 2; ++ni) {
      const bf16* pb = sB + (wn * 32 + ni * 16 + l16) * LDB + lhalf * 8;
      bb[ni] = frag16(pb, pb + 16);
    }
#pragma unroll
    for (int mi = 0; mi < 2; ++mi) {
#pragma unroll
      for (int ni = 0; ni < 2; ++ni) {
        acc[mi][ni] = __builtin_amdgcn_wmma_f32_16x16x32_bf16(
            false, ah[mi], false, bb[ni], (short)0, acc[mi][ni], false, false);
        acc[mi][ni] = __builtin_amdgcn_wmma_f32_16x16x32_bf16(
            false, al[mi], false, bb[ni], (short)0, acc[mi][ni], false, false);
      }
    }
    __syncthreads();
  }

#pragma unroll
  for (int ni = 0; ni < 2; ++ni) {
    const int col = nBase + wn * 32 + ni * 16 + l16;
    const float s = scale[col];
#pragma unroll
    for (int mi = 0; mi < 2; ++mi) {
      const int rbase = mBase + wm * 32 + mi * 16 + (lhalf ? 8 : 0);
#pragma unroll
      for (int r = 0; r < 8; ++r) {
        out[(size_t)(rbase + r) * OUT_F + col] = acc[mi][ni][r] * s;
      }
    }
  }
}

extern "C" void kernel_launch(void* const* d_in, const int* in_sizes, int n_in,
                              void* d_out, int out_size, void* d_ws, size_t ws_size,
                              hipStream_t stream) {
  (void)in_sizes; (void)n_in; (void)out_size;
  const float* x     = (const float*)d_in[0];
  const float* w     = (const float*)d_in[1];
  const float* scale = (const float*)d_in[2];
  float* out = (float*)d_out;

  const size_t x_elems = (size_t)TOKENS * IN_F;
  const size_t w_elems = (size_t)OUT_F * IN_F;
  const size_t need = (x_elems * 2 + w_elems) * sizeof(bf16);  // xhi+xlo+wsign

  dim3 grid(OUT_F / BN, TOKENS / BM);   // 32 x 128 blocks
  dim3 block(256);

  if (d_ws != nullptr && ws_size >= need) {
    bf16* xhi = (bf16*)d_ws;
    bf16* xlo = xhi + x_elems;
    bf16* wsg = xlo + x_elems;
    convert_x_hilo<<<dim3((unsigned)(x_elems / (256 * 4))), block, 0, stream>>>(x, xhi, xlo);
    convert_w_sign<<<dim3((unsigned)(w_elems / (256 * 4))), block, 0, stream>>>(w, wsg);
    fbl_wmma_gemm_async<<<grid, block, 0, stream>>>(xhi, xlo, wsg, scale, out);
  } else {
    fbl_wmma_gemm_fused<<<grid, block, 0, stream>>>(x, w, scale, out);
  }
}
